// MultiHeadSelfAttention_74036646248493
// MI455X (gfx1250) — matrix operand order repaired and checked by reading, not compiled, not run on hardware
//
#include <hip/hip_runtime.h>
#include <hip/hip_bf16.h>

typedef _Float16 half_t;
typedef __attribute__((ext_vector_type(16))) _Float16 v16h;
typedef __attribute__((ext_vector_type(8)))  float    v8f;
typedef __attribute__((ext_vector_type(4)))  unsigned int uint4v;
typedef __attribute__((ext_vector_type(4)))  unsigned int u32x4;
typedef __attribute__((ext_vector_type(8)))  int i32x8;
typedef __attribute__((ext_vector_type(4)))  int i32x4;

#define D_MODEL 1024
#define NUM_HEADS 16
#define D_K 64
#define SEQ 2048
#define BATCH 2
#define NTOK (BATCH * SEQ)   // 4096

#if defined(__gfx1250__) && __has_builtin(__builtin_amdgcn_tensor_load_to_lds)
#define HAVE_TDM 1
#else
#define HAVE_TDM 0
#endif

union V16U { v16h v; uint4v u[2]; };

__device__ __forceinline__ v8f wmma_f16(v16h a, v16h b, v8f c) {
  return __builtin_amdgcn_wmma_f32_16x16x32_f16(
      false, a, false, b, (short)0, c, false, false);
}

// --- WMMA operand loaders (wave32 layouts per CDNA5 ISA 7.12.2) -------------
// A operand (16x32), memory rows M contiguous along K.
__device__ __forceinline__ v16h load_A(const half_t* A, int lda, int m0, int k0, int lane) {
  const half_t* r = A + (size_t)(m0 + (lane & 15)) * lda + k0 + 8 * (lane >> 4);
  V16U u;
  u.u[0] = *(const uint4v*)(r);
  u.u[1] = *(const uint4v*)(r + 16);
  return u.v;
}

// B operand (32x16) from memory laid out [n][k] (k contiguous): B(k,n)=Wn[n*ldw+k].
__device__ __forceinline__ v16h load_Bt(const half_t* Wn, int ldw, int n0, int k0, int lane) {
  const half_t* r = Wn + (size_t)(n0 + (lane & 15)) * ldw + k0 + 8 * (lane >> 4);
  V16U u;
  u.u[0] = *(const uint4v*)(r);
  u.u[1] = *(const uint4v*)(r + 16);
  return u.v;
}

// --- TDM: 2D tile load into LDS (D# per CDNA5 ISA section 8) ----------------
// tensor dims set huge so OOB clipping never fires; stride is the real row pitch.
__device__ __forceinline__ void tdm_load_2d(unsigned lds_off, const void* gsrc,
                                            unsigned stride_d0,
                                            unsigned tile_d0, unsigned tile_d1) {
#if HAVE_TDM
  unsigned long long ga = (unsigned long long)gsrc;
  u32x4 g0;
  g0[0] = 1u;                                         // count=1, user mode
  g0[1] = lds_off;                                    // lds_addr (bytes)
  g0[2] = (unsigned)(ga & 0xffffffffu);               // global_addr[31:0]
  g0[3] = (unsigned)((ga >> 32) & 0x1ffffffu) | (2u << 30);  // [56:32] | type=2
  i32x8 g1;
  const unsigned td0 = 1u << 28, td1 = 1u << 28;      // huge tensor dims (no OOB)
  g1[0] = (int)(1u << 16);                            // wg_mask=0, data_size=1 (2B)
  g1[1] = (int)((td0 & 0xffffu) << 16);               // atomic_addr=0 | dim0.lo16
  g1[2] = (int)(((td0 >> 16) & 0xffffu) | ((td1 & 0xffffu) << 16));
  g1[3] = (int)(((td1 >> 16) & 0xffffu) | ((tile_d0 & 0xffffu) << 16));
  g1[4] = (int)(tile_d1 & 0xffffu);                   // tile_dim1 | tile_dim2=0
  g1[5] = (int)stride_d0;                             // dim0_stride[31:0]
  g1[6] = 0;                                          // stride hi | dim1_stride.lo
  g1[7] = 0;
  i32x4 gz = {0, 0, 0, 0};
#if __clang_major__ >= 23
  i32x8 gz8 = {0, 0, 0, 0, 0, 0, 0, 0};
  __builtin_amdgcn_tensor_load_to_lds(g0, g1, gz, gz, gz8, 0);
#else
  __builtin_amdgcn_tensor_load_to_lds(g0, g1, gz, gz, 0);
#endif
#else
  (void)lds_off; (void)gsrc; (void)stride_d0; (void)tile_d0; (void)tile_d1;
#endif
}

// --- fp32 -> f16 convert ----------------------------------------------------
__global__ void cvt_kernel(const float* __restrict__ src, half_t* __restrict__ dst, int n) {
  int i = blockIdx.x * blockDim.x + threadIdx.x;
  int stride = gridDim.x * blockDim.x;
  for (; i < n; i += stride) dst[i] = (half_t)src[i];
}

// --- V transpose: vt[b][h][d][s] = v[b][s][h*64+d] --------------------------
__global__ void transpose_v_kernel(const half_t* __restrict__ V, half_t* __restrict__ Vt) {
  int i = blockIdx.x * blockDim.x + threadIdx.x;
  if (i >= NTOK * D_MODEL) return;
  int s = i & (SEQ - 1);
  int d = (i >> 11) & (D_K - 1);
  int h = (i >> 17) & (NUM_HEADS - 1);
  int b = i >> 21;
  Vt[(((size_t)(b * NUM_HEADS + h) * D_K) + d) * SEQ + s] =
      V[((size_t)(b * SEQ + s)) * D_MODEL + h * D_K + d];
}

// --- GEMM: D[m,n] = sum_k A[m,k]*W[n,k]; wave = 32x64 of C (8 WMMA accums) --
__global__ __launch_bounds__(256) void gemm_wmma_kernel(
    const half_t* __restrict__ A,
    const half_t* __restrict__ W0, const half_t* __restrict__ W1, const half_t* __restrict__ W2,
    half_t* __restrict__ D0, half_t* __restrict__ D1, half_t* __restrict__ D2,
    float* __restrict__ Df, int storef32)
{
  const int lane = threadIdx.x & 31;
  const int wave = threadIdx.x >> 5;
  const int m0 = blockIdx.x * 256 + wave * 32;
  const int n0 = blockIdx.y * 64;
  const half_t* W = (blockIdx.z == 0) ? W0 : ((blockIdx.z == 1) ? W1 : W2);
  half_t* D = (blockIdx.z == 0) ? D0 : ((blockIdx.z == 1) ? D1 : D2);

  v8f acc[2][4] = {};
  for (int k = 0; k < D_MODEL; k += 32) {
    v16h a0 = load_A(A, D_MODEL, m0,      k, lane);
    v16h a1 = load_A(A, D_MODEL, m0 + 16, k, lane);
    #pragma unroll
    for (int t = 0; t < 4; ++t) {
      v16h bt = load_Bt(W, D_MODEL, n0 + 16 * t, k, lane);
      acc[0][t] = wmma_f16(a0, bt, acc[0][t]);
      acc[1][t] = wmma_f16(a1, bt, acc[1][t]);
    }
  }
  const int n = lane & 15, mh = lane >> 4;
  #pragma unroll
  for (int mi = 0; mi < 2; ++mi)
    #pragma unroll
    for (int t = 0; t < 4; ++t)
      #pragma unroll
      for (int r = 0; r < 8; ++r) {
        size_t off = (size_t)(m0 + 16 * mi + r + 8 * mh) * D_MODEL + n0 + 16 * t + n;
        if (storef32) Df[off] = acc[mi][t][r];
        else          D[off]  = (half_t)acc[mi][t][r];
      }
}

// --- RoPE on Q and K --------------------------------------------------------
__global__ void rope_kernel(half_t* __restrict__ Q, half_t* __restrict__ K,
                            const int* __restrict__ pos)
{
  int i = blockIdx.x * blockDim.x + threadIdx.x;
  if (i >= BATCH * SEQ * NUM_HEADS * (D_K / 2)) return;
  int p = i & 31;
  int h = (i >> 5) & 15;
  int s = (i >> 9) & 2047;
  int b = i >> 20;
  float m = (float)pos[s];
  float freq = __expf(-(float)p * (9.210340371976184f / 32.0f));  // 10000^(-2p/64)
  float sn, c;
  __sincosf(m * freq, &sn, &c);
  size_t base = ((size_t)(b * SEQ + s)) * D_MODEL + h * D_K + 2 * p;
  float q1 = (float)Q[base], q2 = (float)Q[base + 1];
  Q[base]     = (half_t)(q1 * c - q2 * sn);
  Q[base + 1] = (half_t)(q1 * sn + q2 * c);
  float k1 = (float)K[base], k2 = (float)K[base + 1];
  K[base]     = (half_t)(k1 * c - k2 * sn);
  K[base + 1] = (half_t)(k1 * sn + k2 * c);
}

// --- Flash attention --------------------------------------------------------
// Block = 8 waves = 128 query rows of one (b,h); K/Vt tiles staged in LDS once
// per block per 32-key block (TDM tensor_load_to_lds when available).
__global__ __launch_bounds__(256) void attn_kernel(
    const half_t* __restrict__ Q, const half_t* __restrict__ K,
    const half_t* __restrict__ Vt, half_t* __restrict__ O)
{
  __shared__ __align__(32) half_t pshm[8][16 * 32];  // per-wave P relayout
  __shared__ __align__(32) half_t kshm[32 * 64];     // K tile  [key][d]
  __shared__ __align__(32) half_t vshm[64 * 32];     // Vt tile [d][key]
  const int tid  = threadIdx.x;
  const int lane = tid & 31;
  const int wave = tid >> 5;
  const int b = blockIdx.z, h = blockIdx.y;
  const int q0 = blockIdx.x * 128 + wave * 16;
  const half_t* Qb  = Q  + (size_t)b * SEQ * D_MODEL + h * D_K;
  const half_t* Kb  = K  + (size_t)b * SEQ * D_MODEL + h * D_K;
  const half_t* Vtb = Vt + ((size_t)(b * NUM_HEADS + h) * D_K) * SEQ;

  v16h aq0 = load_A(Qb, D_MODEL, q0, 0,  lane);
  v16h aq1 = load_A(Qb, D_MODEL, q0, 32, lane);

  v8f o[4] = {};
  float mrow[8], lrow[8];
  #pragma unroll
  for (int r = 0; r < 8; ++r) { mrow[r] = -3.0e38f; lrow[r] = 0.0f; }

  const int n = lane & 15, mh = lane >> 4;
  const int kb_end = (blockIdx.x * 128 + 127) >> 5;   // block-uniform causal bound
  half_t* pw = &pshm[wave][0];

  for (int kb = 0; kb <= kb_end; ++kb) {
    const int kk = kb * 32;
    __syncthreads();   // previous iteration's LDS reads done before restage
#if HAVE_TDM
    if (wave == 0) {
      tdm_load_2d((unsigned)(unsigned long long)(void*)kshm,
                  Kb + (size_t)kk * D_MODEL, D_MODEL, D_K, 32);   // 32 rows x 64
      tdm_load_2d((unsigned)(unsigned long long)(void*)vshm,
                  Vtb + kk, SEQ, 32, D_K);                        // 64 rows x 32
#if __has_builtin(__builtin_amdgcn_s_wait_tensorcnt)
      __builtin_amdgcn_s_wait_tensorcnt((short)0);
#endif
    }
#else
    {
      int r  = tid >> 3, c  = (tid & 7) * 8;   // K tile: 256 x 16B = 4KB
      *(uint4v*)(kshm + r * 64 + c)  = *(const uint4v*)(Kb + (size_t)(kk + r) * D_MODEL + c);
      int r2 = tid >> 2, c2 = (tid & 3) * 8;   // Vt tile: 256 x 16B = 4KB
      *(uint4v*)(vshm + r2 * 32 + c2) = *(const uint4v*)(Vtb + (size_t)r2 * SEQ + kk + c2);
    }
#endif
    __syncthreads();   // tiles visible to all waves

    // scores: S[16q x 32k] = Q(16x64) * K^T(64x32) from LDS
    v8f s0 = {}, s1 = {};
    s0 = wmma_f16(aq0, load_Bt(kshm, D_K, 0,  0,  lane), s0);
    s0 = wmma_f16(aq1, load_Bt(kshm, D_K, 0,  32, lane), s0);
    s1 = wmma_f16(aq0, load_Bt(kshm, D_K, 16, 0,  lane), s1);
    s1 = wmma_f16(aq1, load_Bt(kshm, D_K, 16, 32, lane), s1);

    #pragma unroll
    for (int r = 0; r < 8; ++r) {
      const int qrow = q0 + r + 8 * mh;
      float v0 = s0[r] * 0.125f;   // 1/sqrt(64)
      float v1 = s1[r] * 0.125f;
      s0[r] = (kk + n      <= qrow) ? v0 : -3.0e38f;
      s1[r] = (kk + 16 + n <= qrow) ? v1 : -3.0e38f;
    }

    #pragma unroll
    for (int r = 0; r < 8; ++r) {
      float mx = fmaxf(s0[r], s1[r]);
      mx = fmaxf(mx, __shfl_xor(mx, 1));
      mx = fmaxf(mx, __shfl_xor(mx, 2));
      mx = fmaxf(mx, __shfl_xor(mx, 4));
      mx = fmaxf(mx, __shfl_xor(mx, 8));
      float mnew = fmaxf(mrow[r], mx);
      float corr = __expf(mrow[r] - mnew);
      float p0 = __expf(s0[r] - mnew);
      float p1 = __expf(s1[r] - mnew);
      s0[r] = p0; s1[r] = p1;
      float rs = p0 + p1;
      rs += __shfl_xor(rs, 1);
      rs += __shfl_xor(rs, 2);
      rs += __shfl_xor(rs, 4);
      rs += __shfl_xor(rs, 8);
      lrow[r] = lrow[r] * corr + rs;
      mrow[r] = mnew;
      #pragma unroll
      for (int t = 0; t < 4; ++t) o[t][r] *= corr;
    }

    // P: C-layout -> A-layout via per-wave LDS bounce
    #pragma unroll
    for (int r = 0; r < 8; ++r) {
      pw[(r + 8 * mh) * 32 + n]      = (half_t)s0[r];
      pw[(r + 8 * mh) * 32 + 16 + n] = (half_t)s1[r];
    }
    __builtin_amdgcn_wave_barrier();
    V16U ua;
    ua.u[0] = *(const uint4v*)(pw + (lane & 15) * 32 + 8 * (lane >> 4));
    ua.u[1] = *(const uint4v*)(pw + (lane & 15) * 32 + 16 + 8 * (lane >> 4));
    v16h pa = ua.v;
    __builtin_amdgcn_wave_barrier();

    // O(16x64) += P(16x32) * V(32x64); B from transposed-V LDS tile
    #pragma unroll
    for (int t = 0; t < 4; ++t)
      o[t] = wmma_f16(pa, load_Bt(vshm, 32, t * 16, 0, lane), o[t]);
  }

  #pragma unroll
  for (int r = 0; r < 8; ++r) {
    float inv = 1.0f / lrow[r];
    #pragma unroll
    for (int t = 0; t < 4; ++t) {
      size_t off = ((size_t)b * SEQ + q0 + r + 8 * mh) * D_MODEL + h * D_K + t * 16 + n;
      O[off] = (half_t)(o[t][r] * inv);
    }
  }
}

extern "C" void kernel_launch(void* const* d_in, const int* in_sizes, int n_in,
                              void* d_out, int out_size, void* d_ws, size_t ws_size,
                              hipStream_t stream) {
  (void)in_sizes; (void)n_in; (void)out_size; (void)ws_size;
  const float* x  = (const float*)d_in[0];
  const float* wq = (const float*)d_in[1];
  const float* wk = (const float*)d_in[2];
  const float* wv = (const float*)d_in[3];
  const float* wo = (const float*)d_in[4];
  const int* tokpos = (const int*)d_in[5];
  float* out = (float*)d_out;

  const size_t NX = (size_t)NTOK * D_MODEL;     // 4,194,304 elems
  const size_t NW = (size_t)D_MODEL * D_MODEL;  // 1,048,576 elems
  half_t* xb   = (half_t*)d_ws;
  half_t* wqb  = xb   + NX;
  half_t* wkb  = wqb  + NW;
  half_t* wvb  = wkb  + NW;
  half_t* wob  = wvb  + NW;
  half_t* qb   = wob  + NW;
  half_t* kbuf = qb   + NX;
  half_t* vb   = kbuf + NX;
  half_t* ab   = vb   + NX;
  half_t* vt   = ab   + NX;   // total ~56 MB of d_ws

  cvt_kernel<<<2048, 256, 0, stream>>>(x,  xb,  (int)NX);
  cvt_kernel<<<1024, 256, 0, stream>>>(wq, wqb, (int)NW);
  cvt_kernel<<<1024, 256, 0, stream>>>(wk, wkb, (int)NW);
  cvt_kernel<<<1024, 256, 0, stream>>>(wv, wvb, (int)NW);
  cvt_kernel<<<1024, 256, 0, stream>>>(wo, wob, (int)NW);

  gemm_wmma_kernel<<<dim3(NTOK / 256, D_MODEL / 64, 3), 256, 0, stream>>>(
      xb, wqb, wkb, wvb, qb, kbuf, vb, nullptr, 0);

  rope_kernel<<<(BATCH * SEQ * NUM_HEADS * (D_K / 2) + 255) / 256, 256, 0, stream>>>(
      qb, kbuf, tokpos);

  transpose_v_kernel<<<(int)(NX + 255) / 256, 256, 0, stream>>>(vb, vt);

  attn_kernel<<<dim3(SEQ / 128, NUM_HEADS, BATCH), 256, 0, stream>>>(qb, kbuf, vt, ab);

  gemm_wmma_kernel<<<dim3(NTOK / 256, D_MODEL / 64, 1), 256, 0, stream>>>(
      ab, wob, wob, wob, nullptr, nullptr, nullptr, out, 1);
}
